// HedgehogAttention_53584011985226
// MI455X (gfx1250) — matrix operand order repaired and checked by reading, not compiled, not run on hardware
//
#include <hip/hip_runtime.h>
#include <hip/hip_bf16.h>

// ---------------------------------------------------------------------------
// Hedgehog attention distillation on gfx1250 (MI455X).
//   outputs (flat): [0] loss, [1 .. 1+B*N*N) pred_attn, [1+B*N*N ..) true_attn
// Dataflow: projections -> bf16 ws; attention kernel does two WMMA passes
// (stats pass, then recompute+normalize+store pass) so the 0.54 GB of output
// is written exactly once (NT stores), never read back.
// ---------------------------------------------------------------------------

#define B_SZ 4
#define N_SZ 4096
#define D_SZ 64
#define F_SZ 128   // 2*d hedgehog feature dim

typedef __attribute__((ext_vector_type(16))) __bf16 v16bf;
typedef __attribute__((ext_vector_type(8)))  __bf16 v8bf;
typedef __attribute__((ext_vector_type(8)))  float  v8f;

// -------------------------- loss init --------------------------------------
__global__ void hh_zero(float* out) {
    if (threadIdx.x == 0) out[0] = 0.0f;
}

// -------------------------- projections ------------------------------------
__device__ inline float dot64(const float4* __restrict__ h,
                              const float4* __restrict__ w) {
    float s = 0.0f;
#pragma unroll
    for (int i = 0; i < 16; ++i) {
        float4 a = h[i], b = w[i];
        s += a.x * b.x + a.y * b.y + a.z * b.z + a.w * b.w;
    }
    return s;
}

// grid = B * (N/64), block = 256 (4 threads per row, 16 outputs each)
__global__ __launch_bounds__(256) void hh_proj(
    const float* __restrict__ hs,
    const float* __restrict__ Wq,  const float* __restrict__ bq,
    const float* __restrict__ Wk,  const float* __restrict__ bk,
    const float* __restrict__ Wqb, const float* __restrict__ Wkb,
    __hip_bfloat16* __restrict__ Q,  __hip_bfloat16* __restrict__ K,
    __hip_bfloat16* __restrict__ QB, __hip_bfloat16* __restrict__ KB)
{
    const int bIdx  = blockIdx.x >> 6;          // / 64
    const int nBase = (blockIdx.x & 63) * 64;
    const int r     = threadIdx.x >> 2;         // 0..63 row in tile
    const int quad  = threadIdx.x & 3;          // 0..3 -> 16 e-values each

    const long row = (long)bIdx * N_SZ + nBase + r;
    const float4* h = (const float4*)(hs + row * D_SZ);
    float4 hreg[16];
#pragma unroll
    for (int i = 0; i < 16; ++i) hreg[i] = h[i];

    for (int el = 0; el < 16; ++el) {
        const int e = quad * 16 + el;
        float zq = bq[e] + dot64(hreg, (const float4*)(Wq  + e * D_SZ));
        float zk = bk[e] + dot64(hreg, (const float4*)(Wk  + e * D_SZ));
        float aq =         dot64(hreg, (const float4*)(Wqb + e * D_SZ));
        float ak =         dot64(hreg, (const float4*)(Wkb + e * D_SZ));

        Q[row * F_SZ + e]          = __float2bfloat16(__expf(zq));
        Q[row * F_SZ + D_SZ + e]   = __float2bfloat16(__expf(-zq));
        K[row * F_SZ + e]          = __float2bfloat16(__expf(zk));
        K[row * F_SZ + D_SZ + e]   = __float2bfloat16(__expf(-zk));
        QB[row * D_SZ + e]         = __float2bfloat16(aq);
        KB[row * D_SZ + e]         = __float2bfloat16(ak);
    }
}

// -------------------------- attention + softmax + loss ---------------------
// Load a WMMA operand: two 16B (8 x bf16) segments -> 16 x bf16 (8 VGPRs).
__device__ inline v16bf ld_op(const __bf16* __restrict__ base, int o1, int o2) {
    v8bf lo = *(const v8bf*)(base + o1);
    v8bf hi = *(const v8bf*)(base + o2);
    return __builtin_shufflevector(lo, hi,
        0, 1, 2, 3, 4, 5, 6, 7, 8, 9, 10, 11, 12, 13, 14, 15);
}

// Compute the two 16x16 score tiles (pred over K=128, true over K=64) for
// column tile j, given this wave's resident A operands.
__device__ inline void score_tiles(
    const __bf16* __restrict__ K, const __bf16* __restrict__ KB,
    const v16bf* __restrict__ aq, const v16bf* __restrict__ aqb,
    long bRow, int j, int ln16, int half, v8f& accP, v8f& accT)
{
    const __bf16* krow  = K  + (bRow + j + ln16) * F_SZ;
    const __bf16* kbrow = KB + (bRow + j + ln16) * D_SZ;

    accP = (v8f){};
#pragma unroll
    for (int c = 0; c < 4; ++c) {
        v16bf bk = ld_op(krow, c * 32 + half * 8, c * 32 + 16 + half * 8);
        accP = __builtin_amdgcn_wmma_f32_16x16x32_bf16(
            false, aq[c], false, bk, (short)0, accP, false, false);
    }
    accT = (v8f){};
#pragma unroll
    for (int c = 0; c < 2; ++c) {
        v16bf bkb = ld_op(kbrow, c * 32 + half * 8, c * 32 + 16 + half * 8);
        accT = __builtin_amdgcn_wmma_f32_16x16x32_bf16(
            false, aqb[c], false, bkb, (short)0, accT, false, false);
    }
    accT *= 0.125f;   // 1/sqrt(d)
}

// grid = B * (N/128), block = 256 (8 waves; each wave owns a 16-row tile)
__global__ __launch_bounds__(256) void hh_attn(
    const __hip_bfloat16* __restrict__ Qp, const __hip_bfloat16* __restrict__ Kp,
    const __hip_bfloat16* __restrict__ QBp, const __hip_bfloat16* __restrict__ KBp,
    float* __restrict__ out)
{
    const __bf16* Q  = reinterpret_cast<const __bf16*>(Qp);
    const __bf16* K  = reinterpret_cast<const __bf16*>(Kp);
    const __bf16* QB = reinterpret_cast<const __bf16*>(QBp);
    const __bf16* KB = reinterpret_cast<const __bf16*>(KBp);

    const int wave = threadIdx.x >> 5;
    const int lane = threadIdx.x & 31;
    const int half = lane >> 4;        // 0: lanes 0-15, 1: lanes 16-31
    const int ln16 = lane & 15;

    const int  bIdx    = blockIdx.x >> 5;             // 32 blocks per batch
    const int  rowBase = (blockIdx.x & 31) * 128 + wave * 16;
    const long bRow    = (long)bIdx * N_SZ;

    // ---- A operands (this wave's 16 rows), per the 16-bit 16x32 A layout:
    //   lane<16: feats {c*32+0..7, c*32+16..23}; lane>=16: +8 within each half
    const __bf16* qrow  = Q  + (bRow + rowBase + ln16) * F_SZ;
    const __bf16* qbrow = QB + (bRow + rowBase + ln16) * D_SZ;
    v16bf aq[4], aqb[2];
#pragma unroll
    for (int c = 0; c < 4; ++c)
        aq[c]  = ld_op(qrow,  c * 32 + half * 8, c * 32 + 16 + half * 8);
#pragma unroll
    for (int c = 0; c < 2; ++c)
        aqb[c] = ld_op(qbrow, c * 32 + half * 8, c * 32 + 16 + half * 8);

    // per-lane online softmax stats: 8 rows (i + 8*half) x my column subset
    float mP[8], sP[8], mT[8], sT[8];
#pragma unroll
    for (int i = 0; i < 8; ++i) {
        mP[i] = -1e30f; sP[i] = 0.0f;
        mT[i] = -1e30f; sT[i] = 0.0f;
    }

    // ================= pass 1: WMMA scores -> online stats only ============
    for (int j = 0; j < N_SZ; j += 16) {
        v8f accP, accT;
        score_tiles(K, KB, aq, aqb, bRow, j, ln16, half, accP, accT);
#pragma unroll
        for (int i = 0; i < 8; ++i) {
            float xp = accP[i], xt = accT[i];
            float mn = fmaxf(mP[i], xp);
            sP[i] = sP[i] * __expf(mP[i] - mn) + __expf(xp - mn);
            mP[i] = mn;
            mn = fmaxf(mT[i], xt);
            sT[i] = sT[i] * __expf(mT[i] - mn) + __expf(xt - mn);
            mT[i] = mn;
        }
    }

    // ---- combine stats across the 16-lane column group (masks < 16) -------
#pragma unroll
    for (int i = 0; i < 8; ++i) {
#pragma unroll
        for (int st = 1; st < 16; st <<= 1) {
            float om = __shfl_xor(mP[i], st, 32);
            float os = __shfl_xor(sP[i], st, 32);
            float mn = fmaxf(mP[i], om);
            sP[i] = sP[i] * __expf(mP[i] - mn) + os * __expf(om - mn);
            mP[i] = mn;
            om = __shfl_xor(mT[i], st, 32);
            os = __shfl_xor(sT[i], st, 32);
            mn = fmaxf(mT[i], om);
            sT[i] = sT[i] * __expf(mT[i] - mn) + os * __expf(om - mn);
            mT[i] = mn;
        }
    }
    float rP[8], rT[8];
#pragma unroll
    for (int i = 0; i < 8; ++i) { rP[i] = 1.0f / sP[i]; rT[i] = 1.0f / sT[i]; }

    float* pred = out + 1 + bRow * N_SZ;
    float* tru  = out + 1 + (long)B_SZ * N_SZ * N_SZ + bRow * N_SZ;

    // ========= pass 2: recompute scores, normalize, NT-store, loss =========
    float lossAcc = 0.0f;
    for (int j = 0; j < N_SZ; j += 16) {
        v8f accP, accT;
        score_tiles(K, KB, aq, aqb, bRow, j, ln16, half, accP, accT);
        // D layout: element i -> row (i + 8*half), this lane -> col (j + ln16)
#pragma unroll
        for (int i = 0; i < 8; ++i) {
            const long idx = (long)(rowBase + i + 8 * half) * N_SZ + j + ln16;
            float p = __expf(accP[i] - mP[i]) * rP[i];
            float t = __expf(accT[i] - mT[i]) * rT[i];
            __builtin_nontemporal_store(p, &pred[idx]);
            __builtin_nontemporal_store(t, &tru[idx]);
            lossAcc -= t * __logf(p + 1e-9f);
        }
    }
#pragma unroll
    for (int st = 16; st >= 1; st >>= 1)
        lossAcc += __shfl_xor(lossAcc, st, 32);
    if (lane == 0)
        atomicAdd(out, lossAcc * (1.0f / (float)(B_SZ * N_SZ)));
}

// ---------------------------------------------------------------------------
extern "C" void kernel_launch(void* const* d_in, const int* in_sizes, int n_in,
                              void* d_out, int out_size, void* d_ws, size_t ws_size,
                              hipStream_t stream) {
    const float* hs  = (const float*)d_in[0];
    const float* Wq  = (const float*)d_in[1];
    const float* bq  = (const float*)d_in[2];
    const float* Wk  = (const float*)d_in[3];
    const float* bk  = (const float*)d_in[4];
    const float* Wqb = (const float*)d_in[5];
    const float* Wkb = (const float*)d_in[6];
    float* out = (float*)d_out;

    // ws layout (bf16): Q[4,4096,128] K[4,4096,128] QB[4,4096,64] KB[4,4096,64]
    __hip_bfloat16* ws = (__hip_bfloat16*)d_ws;
    __hip_bfloat16* Q  = ws;
    __hip_bfloat16* K  = ws + (size_t)B_SZ * N_SZ * F_SZ;
    __hip_bfloat16* QB = K  + (size_t)B_SZ * N_SZ * F_SZ;
    __hip_bfloat16* KB = QB + (size_t)B_SZ * N_SZ * D_SZ;

    hh_zero<<<1, 32, 0, stream>>>(out);
    hh_proj<<<B_SZ * (N_SZ / 64), 256, 0, stream>>>(
        hs, Wq, bq, Wk, bk, Wqb, Wkb, Q, K, QB, KB);
    hh_attn<<<B_SZ * (N_SZ / 128), 256, 0, stream>>>(Q, K, QB, KB, out);
}
